// DeLanNet_inverse_3513283248763
// MI455X (gfx1250) — matrix operand order repaired and checked by reading, not compiled, not run on hardware
//
#include <hip/hip_runtime.h>

typedef __attribute__((ext_vector_type(16))) _Float16 v16h;
typedef __attribute__((ext_vector_type(8)))  float    v8f;

#define DOF 7
#define HID 512
#define NLO 21
#define XW  21            // 3*DOF columns in x
#define WAVES 4
#define TB 16             // samples per wave
#define NTHREADS (WAVES * 32)
#define CH_STRIDE 72      // 64 payload halves + 8 pad (bank spread)

// f16 weight arena offsets (in halves). Zero page at 0.
#define W1PAD 16
#define W2PAD_D 16
#define W2PAD_O 32
#define OFF_W1D 16
#define OFF_W1O (OFF_W1D + HID * W1PAD)
#define OFF_W1G (OFF_W1O + HID * W1PAD)
#define OFF_W2D (OFF_W1G + HID * W1PAD)
#define OFF_W2O (OFF_W2D + HID * W2PAD_D)
#define OFF_W2G (OFF_W2O + HID * W2PAD_O)
#define ARENA_N (OFF_W2G + HID * W2PAD_D)   // 57360 halves

__device__ __forceinline__ v8f wmma16(v16h a, v16h b, v8f c) {
  return __builtin_amdgcn_wmma_f32_16x16x32_f16(false, a, false, b, (short)0, c,
                                                false, false);
}

__device__ __forceinline__ v8f zero8() {
  v8f c;
#pragma unroll
  for (int i = 0; i < 8; ++i) c[i] = 0.0f;
  return c;
}

// Native CDNA5 v_tanh_f32 (TRANS op, co-executes with VALU); libm fallback.
__device__ __forceinline__ float fast_tanh(float x) {
#if __has_builtin(__builtin_amdgcn_tanhf)
  return __builtin_amdgcn_tanhf(x);
#elif __has_builtin(__builtin_amdgcn_tanh_f32)
  return __builtin_amdgcn_tanh_f32(x);
#else
  return tanhf(x);
#endif
}

// Intra-wave LDS handoff: DS ops from one wave complete in order (DScnt), and the
// compiler inserts s_wait_dscnt before consuming loads. We only need a scheduling
// fence to keep program order of may-aliasing DS ops; no hardware barrier.
__device__ __forceinline__ void wave_fence() {
#if __has_builtin(__builtin_amdgcn_wave_barrier)
  __builtin_amdgcn_wave_barrier();
#else
  __syncthreads();
#endif
}

// ---- operand loaders: all unconditional LDS reads (pads/zero-page give 0s) ----

// B operand: W1 tile (K rows 0..15, cols n0..n0+15). W1 stored [col][16] padded.
__device__ __forceinline__ v16h loadB_W1tile(const _Float16* arena, int w1, int n0,
                                             int lane) {
  int col = n0 + (lane & 15);
  int base = (lane < 16) ? (w1 + col * W1PAD) : 0;   // hi lanes: zero page
  v16h b;
#pragma unroll
  for (int k = 0; k < 16; ++k) b[k] = arena[base + k];
  return b;
}

// A operand: W1 rows (7 x hidden K-slice) for c1^T = W1 * e^T. Rows>=7 read pads.
__device__ __forceinline__ v16h loadA_W1rows(const _Float16* arena, int w1, int k0,
                                             int lane) {
  int row = lane & 15;
  bool lo = lane < 16;
  v16h a;
#pragma unroll
  for (int j = 0; j < 8; ++j) {
    int k1 = k0 + (lo ? j : 8 + j);
    int k2 = k0 + (lo ? 16 + j : 24 + j);
    a[j]     = arena[w1 + k1 * W1PAD + row];
    a[8 + j] = arena[w1 + k2 * W1PAD + row];
  }
  return a;
}

// A operand: W2^T (out-rows x hidden K-slice). W2 stored [h][pad]; pad cols are 0.
__device__ __forceinline__ v16h loadA_W2T(const _Float16* arena, int w2, int w2pad,
                                          int rowBase, int k0, int lane) {
  int row = rowBase + (lane & 15);
  bool lo = lane < 16;
  v16h a;
#pragma unroll
  for (int j = 0; j < 8; ++j) {
    int k1 = k0 + (lo ? j : 8 + j);
    int k2 = k0 + (lo ? 16 + j : 24 + j);
    a[j]     = arena[w2 + k1 * w2pad + row];
    a[8 + j] = arena[w2 + k2 * w2pad + row];
  }
  return a;
}

// B operand: W2^T (K=w-index rows, hidden tile cols): contiguous per lane.
__device__ __forceinline__ v16h loadB_W2Ttile(const _Float16* arena, int w2, int w2pad,
                                              int nO, int n0, int lane) {
  int col = n0 + (lane & 15);
  int kb = (lane < 16) ? 0 : 16;
  int base = (kb < nO) ? (w2 + col * w2pad + kb) : 0;  // whole-half invalid -> zero page
  v16h b;
#pragma unroll
  for (int k = 0; k < 16; ++k) b[k] = arena[base + k];
  return b;
}

// A operand: 16 samples x K(=32, 7 valid) slice of staged x (cols 21..23 zero pad).
__device__ __forceinline__ v16h loadA_x(const float* s_x, int wv, int lane, int off) {
  int srow = wv * TB + (lane & 15);
  bool lo = lane < 16;
  v16h a;
#pragma unroll
  for (int j = 0; j < 16; ++j) {
    if (j < DOF) {
      int c = lo ? (off + j) : (XW);       // col 21 is a zero pad
      a[j] = (_Float16)s_x[srow * 24 + c];
    } else {
      a[j] = (_Float16)0.0f;
    }
  }
  return a;
}

// A operand: w matrix (16 samples x K) from [16][32] padded f16 buffer.
__device__ __forceinline__ v16h loadA_w32(const _Float16* wbuf, int lane) {
  int row = lane & 15;
  int jb = (lane < 16) ? 0 : 16;
  v16h a;
#pragma unroll
  for (int j = 0; j < 8; ++j) {
    a[j]     = wbuf[row * 32 + jb + j];
    a[8 + j] = wbuf[row * 32 + 8 + jb + j];
  }
  return a;
}

// B operand: transposed read of a staged [16 samples][32 hidden] f16 chunk.
__device__ __forceinline__ v16h loadB_chunk(const _Float16* chunk, int cb, int lane) {
  int row = lane & 15;
  int base = cb + ((lane < 16) ? 0 : 8);
  v16h b;
#pragma unroll
  for (int k = 0; k < 16; ++k) b[k] = chunk[row * CH_STRIDE + base + (k < 8 ? k : k + 8)];
  return b;
}

// Store a transposed-output C tile (rows = outputs, cols = samples) to [smp][out].
__device__ __forceinline__ void storeCT(float* dst, int stride, v8f c, int outBase,
                                        int nValid, int lane, const float* bias) {
  int smp = lane & 15;
  int add = (lane < 16) ? 0 : 8;
#pragma unroll
  for (int r = 0; r < 8; ++r) {
    int o = outBase + add + r;
    if (o < nValid) dst[smp * stride + o] = c[r] + (bias ? bias[o] : 0.0f);
  }
}

// Phase A: z=tanh(q@W1+b1); t=qDot@W1; h^T += W2^T z^T; dLdot^T += W2^T (s*t)^T
__device__ void phaseA(const _Float16* arena, int w1, int w2, int w2pad,
                       const float* b1, int nO, bool deriv,
                       v16h Aq, v16h Aqd, _Float16* chunk, int lane,
                       v8f& hT0, v8f& hT1, v8f& dT0, v8f& dT1) {
  int smpb = (lane < 16) ? 0 : 8;
  int cidx = lane & 15;
  for (int k0 = 0; k0 < HID; k0 += 32) {
#pragma unroll
    for (int half = 0; half < 2; ++half) {
      int n0 = k0 + 16 * half;
      v16h Bw = loadB_W1tile(arena, w1, n0, lane);
      v8f cz = wmma16(Aq, Bw, zero8());
      v8f ct = zero8();
      if (deriv) ct = wmma16(Aqd, Bw, zero8());
      float bias = b1[n0 + cidx];
      int hc = 16 * half + cidx;
#pragma unroll
      for (int r = 0; r < 8; ++r) {
        float z = fast_tanh(cz[r] + bias);
        chunk[(smpb + r) * CH_STRIDE + hc] = (_Float16)z;
        if (deriv) {
          float s = 1.0f - z * z;
          chunk[(smpb + r) * CH_STRIDE + 32 + hc] = (_Float16)(s * ct[r]);
        }
      }
    }
    wave_fence();
    v16h Bz = loadB_chunk(chunk, 0, lane);
    v16h A0 = loadA_W2T(arena, w2, w2pad, 0, k0, lane);
    hT0 = wmma16(A0, Bz, hT0);
    if (deriv) {
      v16h Bst = loadB_chunk(chunk, 32, lane);
      dT0 = wmma16(A0, Bst, dT0);
      if (nO > 16) {
        v16h A1 = loadA_W2T(arena, w2, w2pad, 16, k0, lane);
        hT1 = wmma16(A1, Bz, hT1);
        dT1 = wmma16(A1, Bst, dT1);
      }
    } else if (nO > 16) {
      v16h A1 = loadA_W2T(arena, w2, w2pad, 16, k0, lane);
      hT1 = wmma16(A1, Bz, hT1);
    }
    wave_fence();
  }
}

// Phase B: a = w@W2^T (per hidden tile); recompute s; e = s*a; c1^T += W1 e^T
__device__ void phaseB(const _Float16* arena, int w1, int w2, int w2pad,
                       const float* b1, int nO,
                       v16h Aq, v16h Aw, _Float16* chunk, int lane, v8f& c1T) {
  int smpb = (lane < 16) ? 0 : 8;
  int cidx = lane & 15;
  for (int k0 = 0; k0 < HID; k0 += 32) {
#pragma unroll
    for (int half = 0; half < 2; ++half) {
      int n0 = k0 + 16 * half;
      v16h Bw1 = loadB_W1tile(arena, w1, n0, lane);
      v8f cz = wmma16(Aq, Bw1, zero8());
      v16h Bw2t = loadB_W2Ttile(arena, w2, w2pad, nO, n0, lane);
      v8f ca = wmma16(Aw, Bw2t, zero8());
      float bias = b1[n0 + cidx];
      int hc = 16 * half + cidx;
#pragma unroll
      for (int r = 0; r < 8; ++r) {
        float z = fast_tanh(cz[r] + bias);
        float s = 1.0f - z * z;
        chunk[(smpb + r) * CH_STRIDE + hc] = (_Float16)(s * ca[r]);
      }
    }
    wave_fence();
    v16h Be = loadB_chunk(chunk, 0, lane);
    v16h Ak = loadA_W1rows(arena, w1, k0, lane);
    c1T = wmma16(Ak, Be, c1T);
    wave_fence();
  }
}

__global__ __launch_bounds__(NTHREADS, 1) void delan_fused(
    const float* __restrict__ x,
    const float* __restrict__ Wd1, const float* __restrict__ bd1,
    const float* __restrict__ Wd2, const float* __restrict__ bd2,
    const float* __restrict__ Wo1, const float* __restrict__ bo1,
    const float* __restrict__ Wo2, const float* __restrict__ bo2,
    const float* __restrict__ Wg1, const float* __restrict__ bg1,
    const float* __restrict__ Wg2, const float* __restrict__ bg2,
    float* __restrict__ out) {
  __shared__ __align__(16) _Float16 arena[ARENA_N];
  __shared__ float s_b1[3][HID];
  __shared__ float s_x[WAVES * TB * 24];
  __shared__ _Float16 s_chunk[WAVES][TB * CH_STRIDE];
  __shared__ float s_hld[WAVES][TB][8];
  __shared__ float s_hlo[WAVES][TB][22];
  __shared__ float s_g  [WAVES][TB][8];
  __shared__ float s_dld[WAVES][TB][8];
  __shared__ float s_dlo[WAVES][TB][22];
  __shared__ float s_c1 [WAVES][TB][8];
  __shared__ __align__(16) _Float16 s_wd[WAVES][TB][32];
  __shared__ __align__(16) _Float16 s_wo[WAVES][TB][32];

  int tid = threadIdx.x;
  int lane = tid & 31;
  int wv = tid >> 5;
  int blockBase = blockIdx.x * (WAVES * TB);
  int sBase = blockBase + wv * TB;
  _Float16* chunk = s_chunk[wv];

  // ---- cooperative staging: zero pads, convert weights to f16, stage x/b1 ----
  {
    unsigned int* az = (unsigned int*)arena;
#pragma unroll 4
    for (int i = tid; i < ARENA_N / 2; i += NTHREADS) az[i] = 0u;
    unsigned int* wz0 = (unsigned int*)&s_wd[0][0][0];
    unsigned int* wz1 = (unsigned int*)&s_wo[0][0][0];
    for (int i = tid; i < WAVES * TB * 16; i += NTHREADS) { wz0[i] = 0u; wz1[i] = 0u; }
  }
  __syncthreads();
  for (int i = tid; i < DOF * HID; i += NTHREADS) {
    int k = i / HID, c = i % HID;
    arena[OFF_W1D + c * W1PAD + k] = (_Float16)Wd1[i];
    arena[OFF_W1O + c * W1PAD + k] = (_Float16)Wo1[i];
    arena[OFF_W1G + c * W1PAD + k] = (_Float16)Wg1[i];
  }
  for (int i = tid; i < HID * DOF; i += NTHREADS) {
    int h = i / DOF, o = i % DOF;
    arena[OFF_W2D + h * W2PAD_D + o] = (_Float16)Wd2[i];
    arena[OFF_W2G + h * W2PAD_D + o] = (_Float16)Wg2[i];
  }
  for (int i = tid; i < HID * NLO; i += NTHREADS) {
    int h = i / NLO, o = i % NLO;
    arena[OFF_W2O + h * W2PAD_O + o] = (_Float16)Wo2[i];
  }
  for (int i = tid; i < HID; i += NTHREADS) {
    s_b1[0][i] = bd1[i];
    s_b1[1][i] = bo1[i];
    s_b1[2][i] = bg1[i];
  }
  for (int i = tid; i < WAVES * TB * 24; i += NTHREADS) {
    int r = i / 24, c = i % 24;
    s_x[i] = (c < XW) ? x[(size_t)(blockBase + r) * XW + c] : 0.0f;
  }
  __syncthreads();

  v16h Aq  = loadA_x(s_x, wv, lane, 0);
  v16h Aqd = loadA_x(s_x, wv, lane, DOF);

  v8f hdT = zero8(), ddT = zero8();
  v8f hoT0 = zero8(), hoT1 = zero8(), doT0 = zero8(), doT1 = zero8();
  v8f hgT = zero8();
  v8f dum0 = zero8(), dum1 = zero8(), dum2 = zero8();

  phaseA(arena, OFF_W1D, OFF_W2D, W2PAD_D, s_b1[0], DOF, true,  Aq, Aqd, chunk, lane,
         hdT, dum0, ddT, dum1);
  phaseA(arena, OFF_W1O, OFF_W2O, W2PAD_O, s_b1[1], NLO, true,  Aq, Aqd, chunk, lane,
         hoT0, hoT1, doT0, doT1);
  phaseA(arena, OFF_W1G, OFF_W2G, W2PAD_D, s_b1[2], DOF, false, Aq, Aqd, chunk, lane,
         hgT, dum0, dum2, dum1);

  storeCT(&s_hld[wv][0][0], 8,  hdT,  0,  DOF, lane, bd2);
  storeCT(&s_hlo[wv][0][0], 22, hoT0, 0,  NLO, lane, bo2);
  storeCT(&s_hlo[wv][0][0], 22, hoT1, 16, NLO, lane, bo2);
  storeCT(&s_g  [wv][0][0], 8,  hgT,  0,  DOF, lane, bg2);
  storeCT(&s_dld[wv][0][0], 8,  ddT,  0,  DOF, lane, nullptr);
  storeCT(&s_dlo[wv][0][0], 22, doT0, 0,  NLO, lane, nullptr);
  storeCT(&s_dlo[wv][0][0], 22, doT1, 16, NLO, lane, nullptr);
  wave_fence();

  float res[DOF];
#pragma unroll
  for (int j = 0; j < DOF; ++j) res[j] = 0.0f;

  if (lane < TB) {  // one sample per lane: 7-dim triangular algebra in fp32
    int srow = wv * TB + lane;
    float qd[DOF], qdd[DOF];
#pragma unroll
    for (int i = 0; i < DOF; ++i) {
      qd[i]  = s_x[srow * 24 + DOF + i];
      qdd[i] = s_x[srow * 24 + 2 * DOF + i];
    }
    float L[DOF][DOF], dLm[DOF][DOF];
#pragma unroll
    for (int i = 0; i < DOF; ++i)
#pragma unroll
      for (int j = 0; j < DOF; ++j) { L[i][j] = 0.0f; dLm[i][j] = 0.0f; }
#pragma unroll
    for (int i = 0; i < DOF; ++i) {
      L[i][i]   = s_hld[wv][lane][i];
      dLm[i][i] = s_dld[wv][lane][i];
#pragma unroll
      for (int j = 0; j < i; ++j) {
        int r = i * (i - 1) / 2 + j;
        L[i][j]   = s_hlo[wv][lane][r];
        dLm[i][j] = s_dlo[wv][lane][r];
      }
    }
    float v[DOF], u1[DOF], tq[DOF];
#pragma unroll
    for (int j = 0; j < DOF; ++j) {
      float av = 0.0f, au = 0.0f, at = 0.0f;
#pragma unroll
      for (int i = 0; i < DOF; ++i) {
        if (i >= j) {
          av += L[i][j] * qd[i];
          au += dLm[i][j] * qd[i];
          at += L[i][j] * qdd[i];
        }
      }
      v[j] = av; u1[j] = au; tq[j] = at;   // v=L^T qd, u1=dL^T qd, tq=L^T qdd
    }
#pragma unroll
    for (int j = 0; j < DOF; ++j) {
      float acc = s_g[wv][lane][j];
#pragma unroll
      for (int mm = 0; mm <= j; ++mm)
        acc += L[j][mm] * (tq[mm] + u1[mm]) + dLm[j][mm] * v[mm];  // tau_m + c2
      res[j] = acc;
    }
#pragma unroll
    for (int i = 0; i < DOF; ++i)
      s_wd[wv][lane][i] = (_Float16)(2.0f * qd[i] * v[i]);
#pragma unroll
    for (int i = 1; i < DOF; ++i)
#pragma unroll
      for (int j = 0; j < i; ++j)
        s_wo[wv][lane][i * (i - 1) / 2 + j] = (_Float16)(2.0f * qd[i] * v[j]);
  }
  wave_fence();

  v16h Awd = loadA_w32(&s_wd[wv][0][0], lane);
  v16h Awo = loadA_w32(&s_wo[wv][0][0], lane);
  v8f c1T = zero8();
  phaseB(arena, OFF_W1D, OFF_W2D, W2PAD_D, s_b1[0], DOF, Aq, Awd, chunk, lane, c1T);
  phaseB(arena, OFF_W1O, OFF_W2O, W2PAD_O, s_b1[1], NLO, Aq, Awo, chunk, lane, c1T);
  storeCT(&s_c1[wv][0][0], 8, c1T, 0, DOF, lane, nullptr);
  wave_fence();

  if (lane < TB) {
    int m = sBase + lane;
#pragma unroll
    for (int j = 0; j < DOF; ++j)
      out[(size_t)m * DOF + j] = res[j] + s_c1[wv][lane][j];
  }
}

extern "C" void kernel_launch(void* const* d_in, const int* in_sizes, int n_in,
                              void* d_out, int out_size, void* d_ws, size_t ws_size,
                              hipStream_t stream) {
  (void)n_in; (void)out_size; (void)d_ws; (void)ws_size;
  const float* x   = (const float*)d_in[0];
  const float* Wd1 = (const float*)d_in[1];
  const float* bd1 = (const float*)d_in[2];
  const float* Wd2 = (const float*)d_in[3];
  const float* bd2 = (const float*)d_in[4];
  const float* Wo1 = (const float*)d_in[5];
  const float* bo1 = (const float*)d_in[6];
  const float* Wo2 = (const float*)d_in[7];
  const float* bo2 = (const float*)d_in[8];
  const float* Wg1 = (const float*)d_in[9];
  const float* bg1 = (const float*)d_in[10];
  const float* Wg2 = (const float*)d_in[11];
  const float* bg2 = (const float*)d_in[12];
  float* out = (float*)d_out;

  int B = in_sizes[0] / XW;                 // 32768
  int blocks = B / (WAVES * TB);            // 512 blocks of 128 threads
  delan_fused<<<blocks, NTHREADS, 0, stream>>>(x, Wd1, bd1, Wd2, bd2,
                                               Wo1, bo1, Wo2, bo2,
                                               Wg1, bg1, Wg2, bg2, out);
}
